// Matching_layer_mm_69355131896548
// MI455X (gfx1250) — matrix operand order repaired and checked by reading, not compiled, not run on hardware
//
#include <hip/hip_runtime.h>
#include <hip/hip_bf16.h>

typedef __attribute__((ext_vector_type(16))) _Float16 v16h;
typedef __attribute__((ext_vector_type(8)))  float    v8f;

#define BB 16
#define CC 512
#define HH 64
#define WW 64
#define PATCH 9
#define BSTRIDE 520   // padded LDS row stride (halfs): 1040B = 65*16B, bank-conflict-free frags

union Frag16 { v16h v; uint4 q[2]; };

__device__ __forceinline__ uint4 ld16(const _Float16* p) {
    return *(const uint4*)p;
}
__device__ __forceinline__ uint4 ld16s(const _Float16* p) {   // LDS
    return *(const uint4*)p;
}
__device__ __forceinline__ int clamp64(int v) {
    return v < 0 ? 0 : (v > 63 ? 63 : v);
}

// Pass 1: L2-normalize over channels, convert to f16, write channel-last [b][y][x][c].
// blockIdx.y selects which tensor (0 -> f1, 1 -> f2).
__global__ __launch_bounds__(256) void norm_kernel(const float* __restrict__ f1,
                                                   const float* __restrict__ f2,
                                                   _Float16* __restrict__ o1,
                                                   _Float16* __restrict__ o2) {
    const int tid = blockIdx.x * 256 + threadIdx.x;     // 0 .. B*H*W-1
    const float*   src = blockIdx.y ? f2 : f1;
    _Float16*      dst = blockIdx.y ? o2 : o1;
    const int b = tid >> 12;          // H*W = 4096
    const int s = tid & 4095;         // y*64 + x
    const float* p = src + (size_t)b * CC * 4096 + s;

    float sum = 0.0f;
#pragma unroll 8
    for (int c = 0; c < CC; ++c) {
        float v = p[(size_t)c * 4096];
        sum = fmaf(v, v, sum);
    }
    const float scale = rsqrtf(sum + 1e-6f);

    _Float16* q = dst + (size_t)tid * CC;
    for (int c = 0; c < CC; c += 8) {
        union { uint4 u; _Float16 h[8]; } pk;
#pragma unroll
        for (int i = 0; i < 8; ++i)
            pk.h[i] = (_Float16)(p[(size_t)(c + i) * 4096] * scale);
        *(uint4*)(q + c) = pk.u;      // 16B packed store
    }
}

// Pass 2: one 9-wave block per (b, x_tile, yrow). The B window (32 f2 columns of
// row yrow) is staged into LDS once and shared by all 9 dh-waves; wave dh computes
// outputs for y = yrow+4-dh via two 16x16 N-tiles of v_wmma_f32_16x16x32_f16 over
// K=512, banded extraction out[m,dw] = G[m][m+dw]. Zero padding applied at
// extraction only (clamped staging addresses), so the K-loop is divergence-free.
__global__ __launch_bounds__(288) void corr_kernel(const _Float16* __restrict__ ws1,
                                                   const _Float16* __restrict__ ws2,
                                                   float* __restrict__ out) {
    const int tid  = threadIdx.x;
    const int lane = tid & 31;
    const int wav  = tid >> 5;            // 0..8 == dh
    const int xt   = blockIdx.x;          // 0..3
    const int yre  = blockIdx.y;          // 0..71
    const int b    = blockIdx.z;
    const int x0   = xt << 4;
    const int xb   = x0 - (PATCH / 2);
    const int yrow = yre - (PATCH / 2);   // f2 row: -4 .. 67
    const bool rowOK = ((unsigned)yrow < (unsigned)HH);

    __shared__ __align__(16) _Float16 Bsh[32][BSTRIDE];   // 32 cols x 512 ch (padded)
    __shared__ float stage[PATCH][16][34];

    // ---- Cooperative stage of the shared B window (32 KB) into LDS ----
    if (rowOK) {
        const _Float16* rowp = ws2 + (((size_t)b * HH + yrow) * WW) * CC;
        for (int idx = tid; idx < 32 * 64; idx += 288) {   // 2048 x 16B chunks
            const int j  = idx >> 6;                       // column 0..31
            const int ch = (idx & 63) << 3;                // channel base 0..504
            const int x  = clamp64(xb + j);                // clamped: garbage cols zeroed later
            *(uint4*)(&Bsh[j][ch]) = ld16(rowp + (size_t)x * CC + ch);
        }
    }
    __syncthreads();

    const int dh = wav;
    const int y  = yrow + (PATCH / 2) - dh;               // output row this wave owns
    const bool yOK = ((unsigned)y < (unsigned)HH);
    if (!yOK) return;                                     // whole-wave uniform exit

    const size_t outBase = (((size_t)b * (PATCH * PATCH)) + (size_t)dh * PATCH) * 4096
                         + (size_t)y * WW + x0;

    if (rowOK) {
        const int m  = lane & 15;
        const int hi = lane >> 4;

        // A fragment rows: f1n pixel (b, y, x0+m), channel-last, from global/L2.
        const _Float16* A = ws1 + ((((size_t)b * HH + y) * WW + x0) + m) * CC;

        // Lane layouts used by this code (contiguous k; A and B use the same order):
        //  A (16x32 f16): lanes 0-15 hold K kk..kk+15 of row m; lanes 16-31 kk+16..kk+31.
        //  B (32x16 f16): lanes 0-15 hold K kk..kk+15 of column n; lanes 16-31 kk+16..kk+31.
        const int koffA = hi * 16;
        const int koffB = hi * 16;
        const _Float16* B0 = &Bsh[m][koffB];              // tile 0: cols xb+0..15
        const _Float16* B1 = &Bsh[16 + m][koffB];         // tile 1: cols xb+16..31

        v8f c0 = {};
        v8f c1 = {};

#pragma unroll 4
        for (int kk = 0; kk < CC; kk += 32) {
            Frag16 a, b0f, b1f;
            a.q[0]   = ld16(A + kk + koffA);
            a.q[1]   = ld16(A + kk + 8 + koffA);
            b0f.q[0] = ld16s(B0 + kk);
            b0f.q[1] = ld16s(B0 + kk + 8);
            b1f.q[0] = ld16s(B1 + kk);
            b1f.q[1] = ld16s(B1 + kk + 8);

            c0 = __builtin_amdgcn_wmma_f32_16x16x32_f16(false, a.v, false, b0f.v,
                                                        (short)0, c0, false, false);
            c1 = __builtin_amdgcn_wmma_f32_16x16x32_f16(false, a.v, false, b1f.v,
                                                        (short)0, c1, false, false);
        }

        // Stage G (16x32) in this wave's slab; D layout: elem r -> row r+8*hi, col lane&15.
#pragma unroll
        for (int r = 0; r < 8; ++r) {
            stage[wav][r + 8 * hi][m]      = c0[r];
            stage[wav][r + 8 * hi][16 + m] = c1[r];
        }
        // LDS ops are in-order within a wave; compiler inserts the dscnt wait.

        for (int i = lane; i < 16 * PATCH; i += 32) {
            const int mm = i / PATCH;
            const int dw = i - mm * PATCH;
            const int xpos = xb + mm + dw;                // f2 x sample
            const bool valid = ((unsigned)xpos < (unsigned)WW);
            const float v = valid ? fmaxf(stage[wav][mm][mm + dw], 0.0f) : 0.0f;
            out[outBase + (size_t)dw * 4096 + mm] = v;
        }
    } else {
        // f2 row out of range: this wave's 144 outputs are all zero (zero padding).
        for (int i = lane; i < 16 * PATCH; i += 32) {
            const int mm = i / PATCH;
            const int dw = i - mm * PATCH;
            out[outBase + (size_t)dw * 4096 + mm] = 0.0f;
        }
    }
}

extern "C" void kernel_launch(void* const* d_in, const int* in_sizes, int n_in,
                              void* d_out, int out_size, void* d_ws, size_t ws_size,
                              hipStream_t stream) {
    const float* f1 = (const float*)d_in[0];
    const float* f2 = (const float*)d_in[1];
    float* out = (float*)d_out;

    const size_t tensorBytes = (size_t)BB * HH * WW * CC * sizeof(_Float16); // 67,108,864
    _Float16* ws1 = (_Float16*)d_ws;
    _Float16* ws2 = (_Float16*)((char*)d_ws + tensorBytes);

    // Pass 1: normalize + f16 channel-last repack for both tensors.
    {
        dim3 grid((BB * HH * WW) / 256, 2, 1);
        norm_kernel<<<grid, 256, 0, stream>>>(f1, f2, ws1, ws2);
    }
    // Pass 2: WMMA correlation; block = (x_tile, yrow+4, b), 9 dh-waves sharing B via LDS.
    {
        dim3 grid(4, HH + 2 * (PATCH / 2), BB);   // 4 x 72 x 16
        corr_kernel<<<grid, 9 * 32, 0, stream>>>(ws1, ws2, out);
    }
}
